// W8A8BFP32OFP32Linear_62697932587023
// MI455X (gfx1250) — 8-bit matrix operand order repaired and checked by reading, compiled, not run on hardware
//
#include <hip/hip_runtime.h>
#include <stdint.h>

typedef int v8i __attribute__((ext_vector_type(8)));

#define M_TOTAL 4096
#define N_TOTAL 4096
#define K_TOTAL 4096
#define BM 256
#define BN 128
#define BK 64
#define LDT 80
#define A_TILE_BYTES (BM * LDT)
#define B_TILE_BYTES (BN * LDT)
#define STAGE_BYTES (A_TILE_BYTES + B_TILE_BYTES)
#define NSTAGES 4
#define NSTEPS (K_TOTAL / BK)

template <int N>
__device__ __forceinline__ void wait_async() {
#if __has_builtin(__builtin_amdgcn_s_wait_asynccnt)
    __builtin_amdgcn_s_wait_asynccnt(N);
#else
    asm volatile("s_wait_asynccnt %c0" :: "i"(N) : "memory");
#endif
}

template <int OFF>
__device__ __forceinline__ void async_cp16(uint32_t lds_addr, uint32_t voff, uint64_t sbase) {
    asm volatile("global_load_async_to_lds_b128 %0, %1, %2 offset:%c3"
                 :: "v"(lds_addr), "v"(voff), "s"(sbase), "i"(OFF) : "memory");
}

__global__ __launch_bounds__(256)
void w8a8_i32_wmma_gemm(const int8_t* __restrict__ X,
                        const int8_t* __restrict__ W,
                        const float*  __restrict__ bias,
                        const float*  __restrict__ alpha_p,
                        float*        __restrict__ out)
{
    __shared__ __align__(16) char lds[NSTAGES * STAGE_BYTES];

    const int tid  = threadIdx.x;
    const int wave = tid >> 5;
    const int lane = tid & 31;
    const int h    = lane >> 4;
    const int l16  = lane & 15;

    const int m_blk = blockIdx.y * BM;
    const int n_blk = blockIdx.x * BN;

    const int wm = (wave >> 1) * 64;
    const int wn = (wave & 1) * 64;

    const uint32_t vA = (uint32_t)(tid * K_TOTAL);
    const int      bRow = tid >> 1;
    const int      bCol = (tid & 1) * 32;
    const uint32_t vB = (uint32_t)(bRow * K_TOTAL + bCol);

    const uint64_t baseA = (uint64_t)(X + (size_t)m_blk * K_TOTAL);
    const uint64_t baseB = (uint64_t)(W + (size_t)n_blk * K_TOTAL);

    const uint32_t ldsBase = (uint32_t)(uintptr_t)&lds[0];
    const uint32_t dA0 = ldsBase + (uint32_t)(tid * LDT);
    const uint32_t dB0 = ldsBase + A_TILE_BYTES + (uint32_t)(bRow * LDT + bCol);

    uint32_t dAbuf[NSTAGES], dBbuf[NSTAGES];
#pragma unroll
    for (int s = 0; s < NSTAGES; ++s) {
        dAbuf[s] = dA0 + s * STAGE_BYTES;
        dBbuf[s] = dB0 + s * STAGE_BYTES;
    }

    auto issue_stage = [&](int ksIdx, int buf) {
        const uint64_t sA = baseA + (uint64_t)(ksIdx * BK);
        const uint64_t sB = baseB + (uint64_t)(ksIdx * BK);
        async_cp16<0>(dAbuf[buf], vA, sA);
        async_cp16<16>(dAbuf[buf], vA, sA);
        async_cp16<32>(dAbuf[buf], vA, sA);
        async_cp16<48>(dAbuf[buf], vA, sA);
        async_cp16<0>(dBbuf[buf], vB, sB);
        async_cp16<16>(dBbuf[buf], vB, sB);
    };

    v8i acc[4][4];
#pragma unroll
    for (int mi = 0; mi < 4; ++mi)
#pragma unroll
        for (int ni = 0; ni < 4; ++ni)
            acc[mi][ni] = (v8i){0, 0, 0, 0, 0, 0, 0, 0};

    auto compute = [&](const char* stage) {
        const char* Ab = stage;
        const char* Bb = stage + A_TILE_BYTES;
        union AF { v8i v; uint2 d[4]; } af[4];
#pragma unroll
        for (int mi = 0; mi < 4; ++mi) {
            const char* base = Ab + (wm + mi * 16 + l16) * LDT + h * 16;
            af[mi].d[0] = *(const uint2*)(base);
            af[mi].d[1] = *(const uint2*)(base + 8);
            af[mi].d[2] = *(const uint2*)(base + 32);
            af[mi].d[3] = *(const uint2*)(base + 40);
        }
        union BF { v8i v; uint4 q[2]; } bf[4];
#pragma unroll
        for (int ni = 0; ni < 4; ++ni) {
            const char* base = Bb + (wn + ni * 16 + l16) * LDT + h * 16;
            bf[ni].q[0] = *(const uint4*)(base);
            bf[ni].q[1] = *(const uint4*)(base + 32);
        }
#pragma unroll
        for (int mi = 0; mi < 4; ++mi)
#pragma unroll
            for (int ni = 0; ni < 4; ++ni)
                acc[mi][ni] = __builtin_amdgcn_wmma_i32_16x16x64_iu8(
                     true, af[mi].v,
                     true, bf[ni].v,
                    acc[mi][ni],
                     false,  false);
    };

    issue_stage(0, 0);
    issue_stage(1, 1);
    issue_stage(2, 2);

    for (int ks = 0; ks < NSTEPS; ks += 4) {
#pragma unroll
        for (int j = 0; j < 4; ++j) {
            if (ks + j + 2 < NSTEPS)      wait_async<12>();
            else if (ks + j + 1 < NSTEPS) wait_async<6>();
            else                          wait_async<0>();
            __syncthreads();
            if (ks + j + 3 < NSTEPS)
                issue_stage(ks + j + 3, (j + 3) & 3);
            compute(lds + j * STAGE_BYTES);
        }
    }

    const float alpha = alpha_p[0];
#pragma unroll
    for (int mi = 0; mi < 4; ++mi) {
#pragma unroll
        for (int ni = 0; ni < 4; ++ni) {
            const int n  = n_blk + wn + ni * 16 + l16;
            const float bv = bias[n];
            const int m0 = m_blk + wm + mi * 16 + h * 8;
            float* po = out + (size_t)m0 * N_TOTAL + n;
#pragma unroll
            for (int v = 0; v < 8; ++v)
                po[(size_t)v * N_TOTAL] = alpha * (float)acc[mi][ni][v] + bv;
        }
    }
}

extern "C" void kernel_launch(void* const* d_in, const int* in_sizes, int n_in,
                              void* d_out, int out_size, void* d_ws, size_t ws_size,
                              hipStream_t stream) {
    const int8_t* x     = (const int8_t*)d_in[0];
    const int8_t* w     = (const int8_t*)d_in[1];
    const float*  bias  = (const float*)d_in[2];
    const float*  alpha = (const float*)d_in[3];
    float*        out   = (float*)d_out;

    dim3 grid(N_TOTAL / BN, M_TOTAL / BM);
    dim3 block(256);
    w8a8_i32_wmma_gemm<<<grid, block, 0, stream>>>(x, w, bias, alpha, out);
}
